// WindowAttention_45629732553121
// MI455X (gfx1250) — matrix operand order repaired and checked by reading, not compiled, not run on hardware
//
#include <hip/hip_runtime.h>

// ---------------------------------------------------------------------------
// Fused window attention for MI455X (gfx1250, wave32, WMMA bf16)
// ---------------------------------------------------------------------------

typedef __attribute__((ext_vector_type(16))) __bf16 v16bf;
typedef __attribute__((ext_vector_type(8)))  float  v8f;

#define BATCH 8
#define CH    128
#define IMH   256
#define IMW   256
#define WSZ   8
#define NWIN  32      // windows per spatial dim
#define NHEAD 4
#define DHEAD 32
#define TOK   64      // tokens per window (8x8)

#define QKV_FRAGS  96             // 24 mt-tiles x 4 k-steps
#define PROJ_FRAGS 32             // 8 mt-tiles x 4 k-steps
#define FRAG_DW    256            // 32 lanes x 8 dwords per 16x32 bf16 fragment

// pack two floats -> dword of two bf16 (lowers to v_cvt_pk_bf16_f32)
__device__ __forceinline__ unsigned pk2bf(float lo, float hi){
  union { unsigned u; __bf16 b[2]; } t;
  t.b[0] = (__bf16)lo;
  t.b[1] = (__bf16)hi;
  return t.u;
}

// A-matrix fragment (16x32 bf16, MxK) from LDS, row-major [M][K], K contiguous.
// Per ISA: lane l (r=l&15,h=l>>4): M=r; dword i: K = (i>>2)*16 + h*8 + (i&3)*2
__device__ __forceinline__ v16bf lds_frag_A(const __bf16* base, int ld){
  int l = threadIdx.x & 31, r = l & 15, h = l >> 4;
  const __bf16* row = base + r * ld;
  union { v16bf v; unsigned u[8]; } t;
#pragma unroll
  for (int i = 0; i < 8; ++i){
    int k = ((i >> 2) << 4) + (h << 3) + ((i & 3) << 1);
    t.u[i] = *(const unsigned*)(row + k);
  }
  return t.v;
}

// B-matrix fragment (32x16 bf16, KxN) from LDS stored as [N][K] (K contiguous).
// Per ISA: lane l: N=l&15; h=l>>4; dword v: K = (v>>2)*16 + h*8 + (v&3)*2
__device__ __forceinline__ v16bf lds_frag_B(const __bf16* base, int ld){
  int l = threadIdx.x & 31, r = l & 15, h = l >> 4;
  const __bf16* row = base + r * ld;
  union { v16bf v; unsigned u[8]; } t;
#pragma unroll
  for (int i = 0; i < 8; ++i){
    int k = ((i >> 2) << 4) + (h << 3) + ((i & 3) << 1);
    t.u[i] = *(const unsigned*)(row + k);
  }
  return t.v;
}

// A-fragment from pre-swizzled workspace: [frag][lane][8 dwords] contiguous.
// Two coalesced b128 loads per lane, no cvt, no perm.
__device__ __forceinline__ v16bf ws_frag(const unsigned* frag){
  int l = threadIdx.x & 31;
  union { v16bf v; uint4 q[2]; } t;
  const uint4* p = (const uint4*)(frag + l * 8);
  t.q[0] = p[0];
  t.q[1] = p[1];
  return t.v;
}

// ---------------------------------------------------------------------------
// Kernel 1: LayerNorm2d statistics: mean/rstd over HxW per (b,c)
// ---------------------------------------------------------------------------
__global__ __launch_bounds__(256) void ln_stats(const float* __restrict__ x,
                                                float* __restrict__ meanp,
                                                float* __restrict__ rstdp){
  __shared__ float s1[256], s2[256];
  const int bc = blockIdx.x;                  // 0 .. B*C-1
  const float* p = x + (size_t)bc * (IMH * IMW);
  float a = 0.f, b = 0.f;
  for (int i = threadIdx.x; i < IMH * IMW; i += 256){
    float v = p[i]; a += v; b += v * v;
  }
  s1[threadIdx.x] = a; s2[threadIdx.x] = b;
  __syncthreads();
  for (int o = 128; o > 0; o >>= 1){
    if (threadIdx.x < o){
      s1[threadIdx.x] += s1[threadIdx.x + o];
      s2[threadIdx.x] += s2[threadIdx.x + o];
    }
    __syncthreads();
  }
  if (threadIdx.x == 0){
    float m = s1[0] * (1.f / (IMH * IMW));
    float v = s2[0] * (1.f / (IMH * IMW)) - m * m;
    meanp[bc] = m;
    rstdp[bc] = rsqrtf(v + 1e-6f);
  }
}

// ---------------------------------------------------------------------------
// Kernel 1b: pre-convert + pre-swizzle f32 weights into WMMA A-fragment order.
// One block per fragment; 256 threads = 32 lanes x 8 dwords.
// ---------------------------------------------------------------------------
__global__ __launch_bounds__(256) void pack_weights(const float* __restrict__ qkv_w,
                                                    const float* __restrict__ proj_w,
                                                    unsigned* __restrict__ wq,
                                                    unsigned* __restrict__ wp){
  const int f    = blockIdx.x;                // 0..127
  const int lane = threadIdx.x & 31;
  const int i    = threadIdx.x >> 5;          // dword index 0..7
  const int r = lane & 15, h = lane >> 4;
  const int k = ((i >> 2) << 4) + (h << 3) + ((i & 3) << 1);

  const float* src;
  unsigned*    dst;
  if (f < QKV_FRAGS){
    int mt = f >> 2, kk = f & 3;
    src = qkv_w + (size_t)(mt * 16 + r) * CH + kk * 32;
    dst = wq + (size_t)f * FRAG_DW;
  } else {
    int g = f - QKV_FRAGS;
    int mt = g >> 2, kk = g & 3;
    src = proj_w + (size_t)(mt * 16 + r) * CH + kk * 32;
    dst = wp + (size_t)g * FRAG_DW;
  }
  dst[lane * 8 + i] = pk2bf(src[k], src[k + 1]);
}

// ---------------------------------------------------------------------------
// Kernel 2: one workgroup per 8x8 window: norm -> qkv GEMM -> attention -> proj
// 256 threads = 8 wave32. All GEMMs via V_WMMA_F32_16X16X32_BF16.
// ---------------------------------------------------------------------------
__global__ __launch_bounds__(256) void win_attn_fused(
    const float* __restrict__ x,
    const float* __restrict__ nw,     const float* __restrict__ nb,
    const unsigned* __restrict__ wq,  const float* __restrict__ qkv_b,
    const unsigned* __restrict__ wp,  const float* __restrict__ proj_b,
    const float* __restrict__ meanp,  const float* __restrict__ rstdp,
    float* __restrict__ out)
{
  __shared__ __bf16 s_xn[TOK * CH];            // 16 KB  normalized window [tok][ch]
  __shared__ __bf16 s_q [NHEAD * TOK * DHEAD]; // 16 KB  [head][tok][d]   (A layout)
  __shared__ __bf16 s_k [NHEAD * TOK * DHEAD]; // 16 KB  [head][tok][d]   (B layout for q.k^T)
  __shared__ __bf16 s_vT[NHEAD * DHEAD * TOK]; // 16 KB  [head][d][tok]   (B layout for P.v)
  __shared__ float  s_sc[TOK * TOK];           // 16 KB  scores f32; P bf16 written in place
  __shared__ __bf16 s_ao[TOK * CH];            // 16 KB  attention out [tok][ch]
  __shared__ float  s_scl[CH], s_sft[CH];      //  1 KB  per-channel affine

  const int tid  = threadIdx.x;
  const int wave = tid >> 5;
  const int lane = tid & 31;
  const int lr = lane & 15, lh = lane >> 4;

  const int win = blockIdx.x;                 // b*1024 + wy*32 + wx
  const int b  = win >> 10;
  const int wy = (win >> 5) & 31;
  const int wx = win & 31;
  const int h0 = wy * WSZ, w0 = wx * WSZ;

  // per-channel affine: xn = x*scl + sft
  if (tid < CH){
    float m  = meanp[b * CH + tid];
    float rs = rstdp[b * CH + tid];
    float sc = rs * nw[tid];
    s_scl[tid] = sc;
    s_sft[tid] = nb[tid] - m * sc;
  }
  __syncthreads();

  // ---- Phase 0: load + normalize window into LDS (2 channels / thread) ----
#pragma unroll 4
  for (int it = 0; it < 16; ++it){
    int idx = it * 256 + tid;                 // 4096 channel-pairs
    int px  = idx & 63;                       // token
    int c0  = (idx >> 6) << 1;                // even channel
    int py  = px >> 3, pxx = px & 7;
    size_t gi = (((size_t)b * CH + c0) * IMH + (h0 + py)) * IMW + (w0 + pxx);
    float v0 = x[gi]                          * s_scl[c0]     + s_sft[c0];
    float v1 = x[gi + (size_t)IMH * IMW]      * s_scl[c0 + 1] + s_sft[c0 + 1];
    *(unsigned*)&s_xn[px * CH + c0] = pk2bf(v0, v1);
  }
  __syncthreads();

  // ---- Phase 1: qkv = qkv_w(384x128) @ xn(128x64) -------------------------
  // 24x4 = 96 output tiles of 16x16, K = 128 in 4 steps. 12 tiles per wave.
  for (int ti = 0; ti < 12; ++ti){
    int t  = wave * 12 + ti;
    int mt = t >> 2, nt = t & 3;
    v8f acc = {};
#pragma unroll
    for (int kk = 0; kk < 4; ++kk){
      v16bf a  = ws_frag(wq + (size_t)(mt * 4 + kk) * FRAG_DW);
      v16bf bb = lds_frag_B(s_xn + (nt * 16) * CH + kk * 32, CH);
      acc = __builtin_amdgcn_wmma_f32_16x16x32_bf16(false, a, false, bb,
                                                    (short)0, acc, false, false);
    }
    int token = nt * 16 + lr;
    int which = mt >> 3;                      // tile-uniform: 0=q 1=k 2=v
    int cb    = (mt & 7) * 16;                // channel base within q/k/v
    if (which == 0){
#pragma unroll
      for (int r = 0; r < 8; ++r){
        int c = cb + r + 8 * lh;
        s_q[((c >> 5) * TOK + token) * DHEAD + (c & 31)] =
            (__bf16)(acc[r] + qkv_b[c]);
      }
    } else if (which == 1){
#pragma unroll
      for (int r = 0; r < 8; ++r){
        int c = cb + r + 8 * lh;
        s_k[((c >> 5) * TOK + token) * DHEAD + (c & 31)] =
            (__bf16)(acc[r] + qkv_b[CH + c]);
      }
    } else {
#pragma unroll
      for (int r = 0; r < 8; ++r){
        int c = cb + r + 8 * lh;
        s_vT[((c >> 5) * DHEAD + (c & 31)) * TOK + token] =
            (__bf16)(acc[r] + qkv_b[2 * CH + c]);
      }
    }
  }
  __syncthreads();

  // ---- Phase 2/3: attention, heads processed sequentially -----------------
  const float scale = 0.17677669529663687f;   // 1/sqrt(32)

  for (int hd = 0; hd < NHEAD; ++hd){
    // scores = (q @ k^T) * scale : M=64,N=64,K=32 -> 4x4 tiles, 2 per wave
    for (int ti = 0; ti < 2; ++ti){
      int t  = wave * 2 + ti;
      int mt = t >> 2, nt = t & 3;
      v16bf a  = lds_frag_A(s_q + (hd * TOK + mt * 16) * DHEAD, DHEAD);
      v16bf bb = lds_frag_B(s_k + (hd * TOK + nt * 16) * DHEAD, DHEAD);
      v8f acc = {};
      acc = __builtin_amdgcn_wmma_f32_16x16x32_bf16(false, a, false, bb,
                                                    (short)0, acc, false, false);
      int n = nt * 16 + lr;
#pragma unroll
      for (int r = 0; r < 8; ++r){
        int m = mt * 16 + r + 8 * lh;
        s_sc[m * TOK + n] = acc[r] * scale;
      }
    }
    __syncthreads();

    // row softmax; write bf16 P in place at the start of each f32 row
    if (tid < TOK){
      float* row = s_sc + tid * TOK;
      float mx = row[0];
      for (int i = 1; i < TOK; ++i) mx = fmaxf(mx, row[i]);
      float sum = 0.f;
      for (int i = 0; i < TOK; ++i){ float e = __expf(row[i] - mx); row[i] = e; sum += e; }
      float inv = 1.f / sum;
      __bf16* prow = (__bf16*)row;            // ascending overwrite is safe
      for (int i = 0; i < TOK; i += 2)
        *(unsigned*)&prow[i] = pk2bf(row[i] * inv, row[i + 1] * inv);
    }
    __syncthreads();

    // o = P @ v : M=64,N=32,K=64 -> 4x2 tiles, 1 per wave, 2 k-steps
    {
      int mt = wave >> 1, nt = wave & 1;
      v8f acc = {};
#pragma unroll
      for (int kk = 0; kk < 2; ++kk){
        // P rows live at pitch 128 bf16 (each f32 row reinterpreted)
        v16bf a  = lds_frag_A((const __bf16*)s_sc + (mt * 16) * (TOK * 2) + kk * 32,
                              TOK * 2);
        v16bf bb = lds_frag_B(s_vT + (hd * DHEAD + nt * 16) * TOK + kk * 32, TOK);
        acc = __builtin_amdgcn_wmma_f32_16x16x32_bf16(false, a, false, bb,
                                                      (short)0, acc, false, false);
      }
      int dd = nt * 16 + lr;
#pragma unroll
      for (int r = 0; r < 8; ++r){
        int token = mt * 16 + r + 8 * lh;
        s_ao[token * CH + hd * DHEAD + dd] = (__bf16)acc[r];
      }
    }
    __syncthreads();
  }

  // ---- Phase 4: out = proj_w(128x128) @ ao(128x64) + proj_b ---------------
  // 8x4 = 32 tiles, 4 per wave, K = 128 in 4 steps. Write directly to HBM.
  for (int ti = 0; ti < 4; ++ti){
    int t  = wave * 4 + ti;
    int mt = t >> 2, nt = t & 3;
    v8f acc = {};
#pragma unroll
    for (int kk = 0; kk < 4; ++kk){
      v16bf a  = ws_frag(wp + (size_t)(mt * 4 + kk) * FRAG_DW);
      v16bf bb = lds_frag_B(s_ao + (nt * 16) * CH + kk * 32, CH);
      acc = __builtin_amdgcn_wmma_f32_16x16x32_bf16(false, a, false, bb,
                                                    (short)0, acc, false, false);
    }
    int px  = nt * 16 + lr;
    int py  = px >> 3, pxx = px & 7;
#pragma unroll
    for (int r = 0; r < 8; ++r){
      int o = mt * 16 + r + 8 * lh;
      size_t gi = (((size_t)b * CH + o) * IMH + (h0 + py)) * IMW + (w0 + pxx);
      out[gi] = acc[r] + proj_b[o];
    }
  }
}

// ---------------------------------------------------------------------------
extern "C" void kernel_launch(void* const* d_in, const int* in_sizes, int n_in,
                              void* d_out, int out_size, void* d_ws, size_t ws_size,
                              hipStream_t stream) {
  const float* x      = (const float*)d_in[0];
  const float* norm_w = (const float*)d_in[1];
  const float* norm_b = (const float*)d_in[2];
  const float* qkv_w  = (const float*)d_in[3];
  const float* qkv_b  = (const float*)d_in[4];
  const float* proj_w = (const float*)d_in[5];
  const float* proj_b = (const float*)d_in[6];
  float* out = (float*)d_out;

  // workspace layout: mean[1024] | rstd[1024] | wq[96*256 dw] | wp[32*256 dw]
  float*    meanp = (float*)d_ws;
  float*    rstdp = meanp + BATCH * CH;
  unsigned* wq    = (unsigned*)(rstdp + BATCH * CH);
  unsigned* wp    = wq + QKV_FRAGS * FRAG_DW;

  ln_stats<<<BATCH * CH, 256, 0, stream>>>(x, meanp, rstdp);
  pack_weights<<<QKV_FRAGS + PROJ_FRAGS, 256, 0, stream>>>(qkv_w, proj_w, wq, wp);
  win_attn_fused<<<BATCH * NWIN * NWIN, 256, 0, stream>>>(
      x, norm_w, norm_b, wq, qkv_b, wp, proj_b, meanp, rstdp, out);
}
